// Seq2Seq_32985348833910
// MI455X (gfx1250) — matrix operand order repaired and checked by reading, not compiled, not run on hardware
//
#include <hip/hip_runtime.h>

// ---------------------------------------------------------------------------
// Types / helpers
// ---------------------------------------------------------------------------
typedef __bf16 bf16;
typedef __attribute__((ext_vector_type(16))) __bf16 v16bf;
typedef __attribute__((ext_vector_type(8)))  __bf16 v8bf;
typedef __attribute__((ext_vector_type(8)))  float  v8f;

__device__ __forceinline__ bf16 f2bf(float f) {
  unsigned u = __float_as_uint(f);
  u += 0x7fffu + ((u >> 16) & 1u);            // round-to-nearest-even
  unsigned short s = (unsigned short)(u >> 16);
  return __builtin_bit_cast(bf16, s);
}
__device__ __forceinline__ float sigm(float x) { return 1.f / (1.f + __expf(-x)); }

// CDNA5 async copy: global -> LDS, 16B per lane, tracked by ASYNCcnt.
__device__ __forceinline__ void async_ld_b128(unsigned lds_off, const void* gaddr) {
  asm volatile("global_load_async_to_lds_b128 %0, %1, off"
               :: "v"(lds_off), "v"(gaddr) : "memory");
}
__device__ __forceinline__ void wait_async0() {
  asm volatile("s_wait_asynccnt 0" ::: "memory");
}

// ---------------------------------------------------------------------------
// bf16 WMMA GEMM:  C[M,N](f32) = A[M,K](bf16) @ W[N,K](bf16)^T + bias, opt tanh
// Requires M%64==0, N%64==0, K%32==0, lda/ldw % 8 == 0 (true for all calls).
// Block = 128 threads = 4 waves; tile 64x64, BK=32, double-buffered LDS filled
// by global_load_async_to_lds_b128. Wave w owns the 32x32 quadrant
// (wr=w>>1, wc=w&1): 2 A-frags x 2 B-frags -> 4 WMMAs per K-step.
// Fragment layouts used by this code (A/B k order is contiguous, not the ISA table order):
//   A 16x32 bf16: lane l holds row m=l&15; elem j -> k=(l>>4)*16+j
//   B 32x16 bf16: lane l holds col n=l&15; elem j -> k=(l>>4)*16+j
//   D 16x16 f32 : lane l, vgpr e -> row=e+(l>>4)*8, col=l&15
// ---------------------------------------------------------------------------
__global__ __launch_bounds__(128) void gemm_bf16_nt(
    const bf16* __restrict__ A, int lda,
    const bf16* __restrict__ W, int ldw,
    const float* __restrict__ bias,
    float* __restrict__ C, int ldc,
    bf16* __restrict__ Cb, int ldcb,
    int M, int N, int K, int act)
{
  __shared__ __align__(32) bf16 As[2][64][48];
  __shared__ __align__(32) bf16 Ws[2][64][48];
  const unsigned BUFSTRIDE = 64u * 48u * 2u;   // 6144 bytes between buffers

  const int tid  = threadIdx.x;
  const int wid  = tid >> 5;        // 0..3
  const int lane = tid & 31;
  const int wr   = wid >> 1;        // 0..1 (row quadrant)
  const int wc   = wid & 1;         // 0..1 (col quadrant)
  const int m16  = lane & 15;
  const int half = lane >> 4;

  const int rowBase = blockIdx.y * 64;
  const int colBase = blockIdx.x * 64;

  v8f acc00 = {}, acc01 = {}, acc10 = {}, acc11 = {};

  // Staging map: 128 threads cover a 64x32 bf16 tile with 2 x 16B chunks each.
  const int ar = tid >> 1;          // 0..63 (tile row / weight row)
  const int ac = (tid & 1) * 8;     // 0 or 8; second chunk at ac+16
  const bf16* Ag = A + (size_t)(rowBase + ar) * lda + ac;
  const bf16* Wg = W + (size_t)(colBase + ar) * ldw + ac;

  const unsigned ldsA0 = (unsigned)(size_t)(const void*)&As[0][ar][ac];
  const unsigned ldsA1 = (unsigned)(size_t)(const void*)&As[0][ar][ac + 16];
  const unsigned ldsW0 = (unsigned)(size_t)(const void*)&Ws[0][ar][ac];
  const unsigned ldsW1 = (unsigned)(size_t)(const void*)&Ws[0][ar][ac + 16];

  // prefetch K-tile k0 into LDS buffer `buf`
  auto prefetch = [&](int k0, int buf) {
    unsigned bo = (unsigned)buf * BUFSTRIDE;
    async_ld_b128(ldsA0 + bo, Ag + k0);
    async_ld_b128(ldsA1 + bo, Ag + k0 + 16);
    async_ld_b128(ldsW0 + bo, Wg + k0);
    async_ld_b128(ldsW1 + bo, Wg + k0 + 16);
  };

  prefetch(0, 0);
  wait_async0();
  __syncthreads();

  int buf = 0;
  for (int k0 = 0; k0 < K; k0 += 32) {
    if (k0 + 32 < K) prefetch(k0 + 32, buf ^ 1);

    // A fragments (rows wr*32 and wr*32+16)
    v16bf a0, a1;
    {
      v8bf lo = *(const v8bf*)&As[buf][wr * 32 + m16][half * 16];
      v8bf hi = *(const v8bf*)&As[buf][wr * 32 + m16][half * 16 + 8];
#pragma unroll
      for (int j = 0; j < 8; ++j) { a0[j] = lo[j]; a0[8 + j] = hi[j]; }
    }
    {
      v8bf lo = *(const v8bf*)&As[buf][wr * 32 + 16 + m16][half * 16];
      v8bf hi = *(const v8bf*)&As[buf][wr * 32 + 16 + m16][half * 16 + 8];
#pragma unroll
      for (int j = 0; j < 8; ++j) { a1[j] = lo[j]; a1[8 + j] = hi[j]; }
    }
    // B fragments (cols wc*32 and wc*32+16); 16 contiguous k per lane
    v16bf b0 = *(const v16bf*)&Ws[buf][wc * 32 + m16][half * 16];
    v16bf b1 = *(const v16bf*)&Ws[buf][wc * 32 + 16 + m16][half * 16];

    acc00 = __builtin_amdgcn_wmma_f32_16x16x32_bf16(false, a0, false, b0,
                                                    (short)0, acc00, false, false);
    acc01 = __builtin_amdgcn_wmma_f32_16x16x32_bf16(false, a0, false, b1,
                                                    (short)0, acc01, false, false);
    acc10 = __builtin_amdgcn_wmma_f32_16x16x32_bf16(false, a1, false, b0,
                                                    (short)0, acc10, false, false);
    acc11 = __builtin_amdgcn_wmma_f32_16x16x32_bf16(false, a1, false, b1,
                                                    (short)0, acc11, false, false);

    wait_async0();        // own async stores to nxt landed
    __syncthreads();      // everyone done reading cur + all nxt data visible
    buf ^= 1;
  }

  const float* bp = bias;
#pragma unroll
  for (int i = 0; i < 2; ++i) {
    const v8f acc_i0 = i ? acc10 : acc00;
    const v8f acc_i1 = i ? acc11 : acc01;
    int r0 = rowBase + wr * 32 + i * 16 + half * 8;
#pragma unroll
    for (int e = 0; e < 8; ++e) {
      int r = r0 + e;
      {
        int c = colBase + wc * 32 + m16;
        float val = acc_i0[e] + (bp ? bp[c] : 0.f);
        if (act) val = tanhf(val);
        C[(size_t)r * ldc + c] = val;
        if (Cb) Cb[(size_t)r * ldcb + c] = f2bf(val);
      }
      {
        int c = colBase + wc * 32 + 16 + m16;
        float val = acc_i1[e] + (bp ? bp[c] : 0.f);
        if (act) val = tanhf(val);
        C[(size_t)r * ldc + c] = val;
        if (Cb) Cb[(size_t)r * ldcb + c] = f2bf(val);
      }
    }
  }
}

// ---------------------------------------------------------------------------
// GRU gate fusion: r=sig(ir+hr), z=sig(iz+hz), n=tanh(in+r*hn), h=(1-z)n+z*h
// ---------------------------------------------------------------------------
__global__ void gru_gate(const float* __restrict__ gi, const float* __restrict__ gh,
                         const float* __restrict__ hprev,
                         float* __restrict__ hf, bf16* __restrict__ hb,
                         bf16* __restrict__ yb, int ldyb,
                         float* __restrict__ yf, int yacc,
                         int BH, int H)
{
  int idx = blockIdx.x * blockDim.x + threadIdx.x;
  if (idx >= BH) return;
  int b = idx / H, c = idx % H;
  const float* gib = gi + (size_t)b * 3 * H;
  const float* ghb = gh + (size_t)b * 3 * H;
  float r = sigm(gib[c] + ghb[c]);
  float z = sigm(gib[H + c] + ghb[H + c]);
  float n = tanhf(gib[2 * H + c] + r * ghb[2 * H + c]);
  float h = (1.f - z) * n + z * hprev[idx];
  hf[idx] = h;
  hb[idx] = f2bf(h);
  if (yb) yb[(size_t)b * ldyb + c] = f2bf(h);
  if (yf) { if (yacc) yf[idx] += h; else yf[idx] = h; }
}

// scores[b][t] = sum_d tanh(q[b][d] + encproj[t*B+b][d]) * v[d]; one wave/(t,b)
__global__ void attn_scores(const float* __restrict__ q, const float* __restrict__ ep,
                            const float* __restrict__ vv, float* __restrict__ scores,
                            int T, int B, int D)
{
  int wid = threadIdx.x >> 5, lane = threadIdx.x & 31;
  int gid = blockIdx.x * 8 + wid;
  if (gid >= T * B) return;
  int t = gid / B, b = gid % B;
  const float* qp = q + (size_t)b * D;
  const float* pp = ep + ((size_t)t * B + b) * D;
  float s = 0.f;
  for (int d = lane; d < D; d += 32) s += tanhf(qp[d] + pp[d]) * vv[d];
#pragma unroll
  for (int off = 16; off; off >>= 1) s += __shfl_xor(s, off, 32);
  if (lane == 0) scores[(size_t)b * T + t] = s;
}

// softmax over T=256 per row b (blockDim.x == 256)
__global__ void softmax256(const float* __restrict__ s, float* __restrict__ a, int T)
{
  __shared__ float red[256];
  int b = blockIdx.x, t = threadIdx.x;
  float x = s[(size_t)b * T + t];
  red[t] = x; __syncthreads();
  for (int off = 128; off; off >>= 1) { if (t < off) red[t] = fmaxf(red[t], red[t + off]); __syncthreads(); }
  float mx = red[0]; __syncthreads();
  float e = __expf(x - mx);
  red[t] = e; __syncthreads();
  for (int off = 128; off; off >>= 1) { if (t < off) red[t] += red[t + off]; __syncthreads(); }
  a[(size_t)b * T + t] = e / red[0];
}

// weighted[b][c] = sum_t a[b][t] * encout[t][b][c]; written bf16 to 2 dests
__global__ void attn_weighted(const float* __restrict__ aw, const float* __restrict__ eo,
                              bf16* __restrict__ d0, int ld0,
                              bf16* __restrict__ d1, int ld1,
                              int T, int B, int H)
{
  int b = blockIdx.x, c = threadIdx.x;   // H=512, 256 threads -> 2 cols each
  float a0 = 0.f, a1 = 0.f;
  for (int t = 0; t < T; ++t) {
    float w = aw[(size_t)b * T + t];
    const float* e = eo + ((size_t)t * B + b) * H;
    a0 += w * e[c];
    a1 += w * e[c + 256];
  }
  d0[(size_t)b * ld0 + c] = f2bf(a0);  d0[(size_t)b * ld0 + c + 256] = f2bf(a1);
  d1[(size_t)b * ld1 + c] = f2bf(a0);  d1[(size_t)b * ld1 + c + 256] = f2bf(a1);
}

__global__ void emb_gather(const float* __restrict__ emb, const int* __restrict__ trg_t,
                           bf16* __restrict__ d0, int ld0,
                           bf16* __restrict__ d1, int ld1, int B, int E)
{
  int idx = blockIdx.x * blockDim.x + threadIdx.x;
  if (idx >= B * E) return;
  int b = idx / E, c = idx % E;
  float v = emb[(size_t)trg_t[b] * E + c];
  d0[(size_t)b * ld0 + c] = f2bf(v);
  d1[(size_t)b * ld1 + c] = f2bf(v);
}

__global__ void cvt_f32_bf16_k(const float* __restrict__ x, bf16* __restrict__ y, int n)
{
  int i = blockIdx.x * blockDim.x + threadIdx.x;
  if (i < n) y[i] = f2bf(x[i]);
}

__global__ void add2_bf16_k(const float* __restrict__ a, const float* __restrict__ b,
                            bf16* __restrict__ o, int n)
{
  int i = blockIdx.x * blockDim.x + threadIdx.x;
  if (i < n) o[i] = f2bf(a[i] + b[i]);
}

__global__ void zero_u32(unsigned* __restrict__ p, int n)
{
  int i = blockIdx.x * blockDim.x + threadIdx.x;
  if (i < n) p[i] = 0u;
}

// ---------------------------------------------------------------------------
// Orchestration
// ---------------------------------------------------------------------------
extern "C" void kernel_launch(void* const* d_in, const int* in_sizes, int n_in,
                              void* d_out, int out_size, void* d_ws, size_t ws_size,
                              hipStream_t stream)
{
  (void)in_sizes; (void)n_in; (void)out_size; (void)ws_size;
  const int T = 256, TT = 64, B = 128, H = 512, EMB = 256, V = 8192;

  // ---- inputs (setup_inputs() dict order, params flattened in insertion order)
  const float* src = (const float*)d_in[0];
  const int*   trg = (const int*)d_in[1];
  const float *eWi[2][2], *eWh[2][2], *ebi[2][2], *ebh[2][2];
  int p = 2;
  for (int l = 0; l < 2; ++l)
    for (int d = 0; d < 2; ++d) {
      eWi[l][d] = (const float*)d_in[p++]; eWh[l][d] = (const float*)d_in[p++];
      ebi[l][d] = (const float*)d_in[p++]; ebh[l][d] = (const float*)d_in[p++];
    }
  const float* fcW  = (const float*)d_in[p++]; const float* fcb  = (const float*)d_in[p++];
  const float* attW = (const float*)d_in[p++]; const float* attb = (const float*)d_in[p++];
  const float* vvec = (const float*)d_in[p++]; const float* emb  = (const float*)d_in[p++];
  const float *dWi[2], *dWh[2], *dbi[2], *dbh[2];
  for (int l = 0; l < 2; ++l) {
    dWi[l] = (const float*)d_in[p++]; dWh[l] = (const float*)d_in[p++];
    dbi[l] = (const float*)d_in[p++]; dbh[l] = (const float*)d_in[p++];
  }
  const float* foW = (const float*)d_in[p++]; const float* fob = (const float*)d_in[p++];

  // ---- workspace carve-out
  size_t cur = 0;
  auto alloc = [&](size_t bytes) -> void* {
    cur = (cur + 255) & ~(size_t)255;
    void* q = (char*)d_ws + cur;
    cur += bytes;
    return q;
  };
  bf16* srcb   = (bf16*)alloc((size_t)T * B * 512 * 2);
  bf16* xcat   = (bf16*)alloc((size_t)T * B * 1024 * 2);
  float* GI    = (float*)alloc((size_t)T * B * 1536 * 4);   // reused as encproj
  float* encproj = GI;                                      // [T*B,512] after encoder
  float* encoutf = (float*)alloc((size_t)T * B * 512 * 4);
  bf16*  encoutb = (bf16*)alloc((size_t)T * B * 512 * 2);
  float* gh_s  = (float*)alloc((size_t)B * 1536 * 4);
  float* gi_s  = (float*)alloc((size_t)B * 1536 * 4);
  float* hFf   = (float*)alloc((size_t)B * H * 4);
  float* hBf   = (float*)alloc((size_t)B * H * 4);
  bf16*  hFb   = (bf16*)alloc((size_t)B * H * 2);
  bf16*  hBb   = (bf16*)alloc((size_t)B * H * 2);
  bf16*  hsum_l  = (bf16*)alloc((size_t)B * H * 2);
  float* hiddenf = (float*)alloc((size_t)2 * B * H * 4);
  bf16*  hidb    = (bf16*)alloc((size_t)2 * B * H * 2);
  bf16*  hsumd   = (bf16*)alloc((size_t)B * H * 2);
  float* qf      = (float*)alloc((size_t)B * H * 4);
  float* scores  = (float*)alloc((size_t)B * T * 4);
  float* attwt   = (float*)alloc((size_t)B * T * 4);
  bf16*  cat0    = (bf16*)alloc((size_t)B * 768 * 2);
  bf16*  foin    = (bf16*)alloc((size_t)B * 1280 * 2);
  // bf16 weights
  bf16* eWib[2][2]; bf16* eWhb[2][2];
  for (int l = 0; l < 2; ++l)
    for (int d = 0; d < 2; ++d) {
      eWib[l][d] = (bf16*)alloc((size_t)1536 * (l ? 1024 : 512) * 2);
      eWhb[l][d] = (bf16*)alloc((size_t)1536 * 512 * 2);
    }
  bf16* fcWb  = (bf16*)alloc((size_t)512 * 512 * 2);
  bf16* attWb = (bf16*)alloc((size_t)512 * 1024 * 2);
  bf16* dWib[2]; bf16* dWhb[2];
  dWib[0] = (bf16*)alloc((size_t)1536 * 768 * 2);
  dWhb[0] = (bf16*)alloc((size_t)1536 * 512 * 2);
  dWib[1] = (bf16*)alloc((size_t)1536 * 512 * 2);
  dWhb[1] = (bf16*)alloc((size_t)1536 * 512 * 2);
  bf16* foWb = (bf16*)alloc((size_t)V * 1280 * 2);

  auto cvt = [&](const float* s, bf16* d, size_t n) {
    int ni = (int)n;
    cvt_f32_bf16_k<<<dim3((ni + 255) / 256), 256, 0, stream>>>(s, d, ni);
  };
  auto gemm = [&](const bf16* A, int lda, const bf16* W, int ldw, const float* bias,
                  float* C, int ldc, bf16* Cb, int ldcb, int M, int N, int K, int act) {
    dim3 g(N / 64, M / 64);
    gemm_bf16_nt<<<g, 128, 0, stream>>>(A, lda, W, ldw, bias, C, ldc, Cb, ldcb, M, N, K, act);
  };
  auto zero = [&](void* ptr, size_t words) {
    int ni = (int)words;
    zero_u32<<<dim3((ni + 255) / 256), 256, 0, stream>>>((unsigned*)ptr, ni);
  };

  // ---- one-time conversions
  for (int l = 0; l < 2; ++l)
    for (int d = 0; d < 2; ++d) {
      cvt(eWi[l][d], eWib[l][d], (size_t)1536 * (l ? 1024 : 512));
      cvt(eWh[l][d], eWhb[l][d], (size_t)1536 * 512);
    }
  cvt(fcW, fcWb, 512 * 512);
  cvt(attW, attWb, (size_t)512 * 1024);
  cvt(dWi[0], dWib[0], (size_t)1536 * 768);
  cvt(dWh[0], dWhb[0], (size_t)1536 * 512);
  cvt(dWi[1], dWib[1], (size_t)1536 * 512);
  cvt(dWh[1], dWhb[1], (size_t)1536 * 512);
  cvt(foW, foWb, (size_t)V * 1280);
  cvt(src, srcb, (size_t)T * B * 512);

  // ---- encoder
  for (int l = 0; l < 2; ++l) {
    const int din = l ? 1024 : 512;
    const bf16* X = l ? xcat : srcb;
    for (int d = 0; d < 2; ++d) {
      // input-side gates for all T in one big WMMA GEMM
      gemm(X, din, eWib[l][d], din, ebi[l][d], GI, 1536, nullptr, 0, T * B, 1536, din, 0);
      float* hf_ = d ? hBf : hFf;
      bf16*  hb_ = d ? hBb : hFb;
      zero(hf_, (size_t)B * H);          // f32 words
      zero(hb_, (size_t)B * H / 2);      // bf16 as u32 words
      for (int s = 0; s < T; ++s) {
        int t = d ? (T - 1 - s) : s;
        gemm(hb_, 512, eWhb[l][d], 512, ebh[l][d], gh_s, 1536, nullptr, 0, B, 1536, 512, 0);
        bf16* yb = nullptr; int ldyb = 0; float* yf = nullptr; int yacc = 0;
        if (l == 0) { yb = xcat + (size_t)t * B * 1024 + (size_t)d * 512; ldyb = 1024; }
        else        { yf = encoutf + (size_t)t * B * 512; yacc = d; }
        gru_gate<<<dim3(256), 256, 0, stream>>>(GI + (size_t)t * B * 1536, gh_s, hf_,
                                                hf_, hb_, yb, ldyb, yf, yacc, B * H, H);
      }
    }
    // hidden[l] = tanh((hf+hb) @ fcW.T + fcb)
    add2_bf16_k<<<dim3(256), 256, 0, stream>>>(hFf, hBf, hsum_l, B * H);
    gemm(hsum_l, 512, fcWb, 512, fcb, hiddenf + (size_t)l * B * H, 512,
         hidb + (size_t)l * B * H, 512, B, 512, 512, 1);
  }

  // ---- enc_proj = encout @ We_a.T + attnb   (We_a = attW[:,512:], ldw=1024)
  cvt(encoutf, encoutb, (size_t)T * B * 512);
  gemm(encoutb, 512, attWb + 512, 1024, attb, encproj, 512, nullptr, 0, T * B, 512, 512, 0);

  // ---- decoder
  for (int t = 0; t < TT; ++t) {
    add2_bf16_k<<<dim3(256), 256, 0, stream>>>(hiddenf, hiddenf + (size_t)B * H, hsumd, B * H);
    gemm(hsumd, 512, attWb, 1024, nullptr, qf, 512, nullptr, 0, B, 512, 512, 0);   // Wh_a
    attn_scores<<<dim3((T * B) / 8), 256, 0, stream>>>(qf, encproj, vvec, scores, T, B, 512);
    softmax256<<<dim3(B), 256, 0, stream>>>(scores, attwt, T);
    attn_weighted<<<dim3(B), 256, 0, stream>>>(attwt, encoutf, cat0 + 256, 768,
                                               foin + 512, 1280, T, B, 512);
    emb_gather<<<dim3((B * EMB) / 256), 256, 0, stream>>>(emb, trg + (size_t)t * B,
                                                          cat0, 768, foin + 1024, 1280, B, EMB);
    // layer 0
    gemm(cat0, 768, dWib[0], 768, dbi[0], gi_s, 1536, nullptr, 0, B, 1536, 768, 0);
    gemm(hidb, 512, dWhb[0], 512, dbh[0], gh_s, 1536, nullptr, 0, B, 1536, 512, 0);
    gru_gate<<<dim3(256), 256, 0, stream>>>(gi_s, gh_s, hiddenf, hiddenf, hidb,
                                            nullptr, 0, nullptr, 0, B * H, H);
    // layer 1 (input = new h0)
    gemm(hidb, 512, dWib[1], 512, dbi[1], gi_s, 1536, nullptr, 0, B, 1536, 512, 0);
    gemm(hidb + (size_t)B * H, 512, dWhb[1], 512, dbh[1], gh_s, 1536, nullptr, 0, B, 1536, 512, 0);
    gru_gate<<<dim3(256), 256, 0, stream>>>(gi_s, gh_s, hiddenf + (size_t)B * H,
                                            hiddenf + (size_t)B * H, hidb + (size_t)B * H,
                                            foin, 1280, nullptr, 0, B * H, H);
    // pred = cat(h1, weighted, emb) @ foW.T + fob -> out[b][t][:]
    gemm(foin, 1280, foWb, 1280, fob, (float*)d_out + (size_t)t * V, TT * V,
         nullptr, 0, B, V, 1280, 0);
  }
}
